// Cross_SwinTransformerBlock_3238405341251
// MI455X (gfx1250) — matrix operand order repaired and checked by reading, not compiled, not run on hardware
//
#include <hip/hip_runtime.h>
#include <hip/hip_bf16.h>
#include <math.h>

typedef __attribute__((ext_vector_type(16))) _Float16 v16h;
typedef __attribute__((ext_vector_type(8)))  float    v8f;
typedef __attribute__((ext_vector_type(4)))  unsigned int v4u;
typedef __attribute__((ext_vector_type(8)))  int          v8i;
typedef __attribute__((ext_vector_type(4)))  int          v4i;

#define C_     96
#define NH_    6
#define HD_    16
#define SS_    4
#define MTOT   131072          // B * H * W = 8*128*128
#define NWINS  2048            // B * 256 windows
#define MSTRIP 16              // M-tiles per wave (256 rows)

#if defined(__has_builtin)
#if __has_builtin(__builtin_amdgcn_tensor_load_to_lds)
#define HAVE_TDM 1
#else
#define HAVE_TDM 0
#endif
#else
#define HAVE_TDM 0
#endif

// ---------------------------------------------------------------------------
// TDM: one tensor_load_to_lds for a 2-D tile (D# packing per ISA §8.3/§8.4).
// ---------------------------------------------------------------------------
__device__ __forceinline__ void tdm_load_2d(unsigned lds_off, const void* gptr,
                                            int dsz, unsigned tdim0, unsigned tdim1,
                                            unsigned stride0, unsigned tw, unsigned th) {
#if HAVE_TDM
  unsigned long long ga = (unsigned long long)gptr;
  v4u g0;
  g0[0] = 1u;                                                // count=1, user descriptor
  g0[1] = lds_off;                                           // lds_addr
  g0[2] = (unsigned)(ga & 0xFFFFFFFFu);                      // global_addr lo
  g0[3] = (unsigned)((ga >> 32) & 0x01FFFFFFu) | (2u << 30); // addr hi | type=2
  v8i g1;
  g1[0] = (int)((unsigned)dsz << 16);                        // data_size
  g1[1] = (int)((tdim0 & 0xFFFFu) << 16);                    // tensor_dim0 lo16
  g1[2] = (int)(((tdim0 >> 16) & 0xFFFFu) | ((tdim1 & 0xFFFFu) << 16));
  g1[3] = (int)(((tdim1 >> 16) & 0xFFFFu) | ((tw & 0xFFFFu) << 16)); // tile_dim0
  g1[4] = (int)(th & 0xFFFFu);                               // tile_dim1
  g1[5] = (int)stride0;                                      // tensor_dim0_stride
  g1[6] = 0;
  g1[7] = 0;
  v4i z4 = {0, 0, 0, 0};
#if __clang_major__ >= 23
  v8i z8 = {0, 0, 0, 0, 0, 0, 0, 0};
  __builtin_amdgcn_tensor_load_to_lds(g0, g1, z4, z4, z8, 0);
#else
  __builtin_amdgcn_tensor_load_to_lds(g0, g1, z4, z4, 0);
#endif
#endif
}

__device__ __forceinline__ void tdm_wait() {
#if defined(__has_builtin)
#if __has_builtin(__builtin_amdgcn_s_wait_tensorcnt)
  __builtin_amdgcn_s_wait_tensorcnt(0);
#endif
#endif
}

// ---------------------------------------------------------------------------
// WMMA helper: D = A(16x32 f16) * B(32x16 f16) + C(f32)
// ---------------------------------------------------------------------------
__device__ __forceinline__ v8f wmma32(v16h a, v16h b, v8f c) {
  return __builtin_amdgcn_wmma_f32_16x16x32_f16(
      false, a, false, b, (short)0, c, false, false);
}

// Build all K/32 B-fragments in registers from the staged f32 weight tile.
// Done ONCE per block; invariant across the 16-tile M-strip.
template <int K>
__device__ __forceinline__ void load_bfrag(v16h* bf, const float* Ws, int lane) {
  const int n   = lane & 15;
  const int kbB = (lane < 16) ? 0 : 16;   // B frag: lo lanes K 0-15, hi 16-31
#pragma unroll
  for (int kk = 0; kk < K / 32; ++kk) {
#pragma unroll
    for (int v = 0; v < 8; ++v) {
      int k = kk * 32 + kbB + 2 * v;
      bf[kk][2 * v]     = (_Float16)Ws[k * 16 + n];
      bf[kk][2 * v + 1] = (_Float16)Ws[(k + 1) * 16 + n];
    }
  }
}

// One 16x16 output tile from an LDS A tile and register B fragments.
template <int K>
__device__ __forceinline__ v8f gemm_tile_reg(const _Float16* As, const v16h* bf,
                                             int lane) {
  v8f acc = {};
  const int m   = lane & 15;
  const int kbA = (lane < 16) ? 0 : 16;   // A frag: lo lanes K 0-15, hi 16-31
#pragma unroll
  for (int kk = 0; kk < K / 32; ++kk) {
    const _Float16* ap = As + m * K + kk * 32;
    v16h af;
#pragma unroll
    for (int v = 0; v < 8; ++v) {
      int k = kbA + 2 * v;
      af[2 * v]     = ap[k];
      af[2 * v + 1] = ap[k + 1];
    }
    acc = wmma32(af, bf[kk], acc);
  }
  return acc;
}

// Branch-free exact-form GELU: Abramowitz–Stegun erf, |err| < 1.5e-7.
__device__ __forceinline__ float gelu_erf(float x) {
  float z  = x * 0.70710678118654752f;
  float az = fabsf(z);
  float t  = __builtin_amdgcn_rcpf(1.f + 0.3275911f * az);
  float p  = t * (0.254829592f +
             t * (-0.284496736f +
             t * (1.421413741f +
             t * (-1.453152027f +
             t * 1.061405429f))));
  float e  = __expf(-az * az);
  float er = 1.f - p * e;
  er = (z < 0.f) ? -er : er;
  return 0.5f * x * (1.f + er);
}

// ---------------------------------------------------------------------------
// K1: LayerNorm + roll(-SS) + window-partition, f32 -> f16 windowed layout
// ---------------------------------------------------------------------------
__global__ void k_ln_window(const float* __restrict__ x,
                            const float* __restrict__ g,
                            const float* __restrict__ b,
                            _Float16* __restrict__ xw) {
  int t = blockIdx.x * blockDim.x + threadIdx.x;
  if (t >= MTOT) return;
  int win = t >> 6, tok = t & 63;
  int bt = win >> 8, wi = win & 255;
  int wh = wi >> 4, wc = wi & 15;
  int th = tok >> 3, tw = tok & 7;
  int h = (wh * 8 + th + SS_) & 127;
  int w = (wc * 8 + tw + SS_) & 127;
  const float* row = x + ((size_t)bt * 16384 + h * 128 + w) * C_;
  __builtin_prefetch(row, 0, 1);
  float mu = 0.f;
#pragma unroll 4
  for (int c = 0; c < C_; ++c) mu += row[c];
  mu *= (1.f / C_);
  float var = 0.f;
#pragma unroll 4
  for (int c = 0; c < C_; ++c) { float d = row[c] - mu; var += d * d; }
  var *= (1.f / C_);
  float rs = rsqrtf(var + 1e-5f);
  _Float16* o = xw + (size_t)t * C_;
#pragma unroll 4
  for (int c = 0; c < C_; ++c)
    o[c] = (_Float16)((row[c] - mu) * rs * g[c] + b[c]);
}

__global__ void k_ln_plain(const float* __restrict__ x,
                           const float* __restrict__ g,
                           const float* __restrict__ b,
                           _Float16* __restrict__ o16) {
  int t = blockIdx.x * blockDim.x + threadIdx.x;
  if (t >= MTOT) return;
  const float* row = x + (size_t)t * C_;
  __builtin_prefetch(row, 0, 1);
  float mu = 0.f;
#pragma unroll 4
  for (int c = 0; c < C_; ++c) mu += row[c];
  mu *= (1.f / C_);
  float var = 0.f;
#pragma unroll 4
  for (int c = 0; c < C_; ++c) { float d = row[c] - mu; var += d * d; }
  var *= (1.f / C_);
  float rs = rsqrtf(var + 1e-5f);
  _Float16* o = o16 + (size_t)t * C_;
#pragma unroll 4
  for (int c = 0; c < C_; ++c)
    o[c] = (_Float16)((row[c] - mu) * rs * g[c] + b[c]);
}

// ---------------------------------------------------------------------------
// K2: GEMM strip kernel (K compile-time). One wave owns a 256-row M-strip for
//     one 16-col slab: W tile staged via TDM, converted to register B-frags
//     once; A tiles double-buffered via TDM overlapping the WMMA k-loop.
//     O_f16 = scale * (A_f16 @ W_f32 + bias), optional exact GELU.
// ---------------------------------------------------------------------------
template <int K>
__global__ void k_gemm_f16(const _Float16* __restrict__ Aa,
                           const float* __restrict__ Wt,
                           const float* __restrict__ bias,
                           _Float16* __restrict__ O,
                           int N, float scale, int gelu) {
  constexpr int WBYTES = K * 16 * 4;
  constexpr int ABYTES = 16 * K * 2;
  __shared__ __align__(16) char smem[WBYTES + 2 * ABYTES];
  float*    Ws = (float*)smem;
  _Float16* A0 = (_Float16*)(smem + WBYTES);
  _Float16* A1 = (_Float16*)(smem + WBYTES + ABYTES);
  int ncols = N >> 4;
  int tm0 = (blockIdx.x / ncols) * MSTRIP;
  int tn  = blockIdx.x % ncols;
  int lane = threadIdx.x;
  const int n  = (lane & 15) + tn * 16;
  const int mb = (lane < 16) ? 0 : 8;
  float bv = bias[n];
#if HAVE_TDM
  tdm_load_2d((unsigned)(uintptr_t)Ws, Wt + tn * 16, 2, N, K, N, 16, K);
  tdm_load_2d((unsigned)(uintptr_t)A0, Aa + (size_t)tm0 * 16 * K, 1, K, 16, K, K, 16);
  tdm_wait();
#else
  for (int i = lane; i < K * 16; i += 32)
    Ws[i] = Wt[(size_t)(i >> 4) * N + tn * 16 + (i & 15)];
  __syncthreads();
#endif
  v16h bf[K / 32];
  load_bfrag<K>(bf, Ws, lane);          // once per block; registers thereafter
  for (int i = 0; i < MSTRIP; ++i) {
#if HAVE_TDM
    const _Float16* Acur = (i & 1) ? A1 : A0;
    _Float16*       Anxt = (i & 1) ? A0 : A1;
    if (i + 1 < MSTRIP)
      tdm_load_2d((unsigned)(uintptr_t)Anxt,
                  Aa + (size_t)(tm0 + i + 1) * 16 * K, 1, K, 16, K, K, 16);
#else
    const _Float16* Acur = A0;
    for (int idx = lane; idx < 16 * K; idx += 32)
      A0[idx] = Aa[(size_t)(tm0 + i) * 16 * K + idx];
    __syncthreads();
#endif
    v8f acc = gemm_tile_reg<K>(Acur, bf, lane);
#pragma unroll
    for (int r = 0; r < 8; ++r) {
      float v = acc[r] + bv;
      if (gelu) v = gelu_erf(v);
      O[(size_t)((tm0 + i) * 16 + mb + r) * N + n] = (_Float16)(v * scale);
    }
#if HAVE_TDM
    tdm_wait();
#else
    __syncthreads();
#endif
  }
}

// ---------------------------------------------------------------------------
// K3: windowed cross-attention, one wave per (window, head).
// ---------------------------------------------------------------------------
__global__ void k_attn(const _Float16* __restrict__ q,
                       const _Float16* __restrict__ kv,
                       const float* __restrict__ rpb,
                       _Float16* __restrict__ O) {
  __shared__ __align__(16) char smem[16384 + 8192 + 3 * 2048];
  float*    S  = (float*)smem;
  _Float16* P  = (_Float16*)(smem + 16384);
  _Float16* qs = (_Float16*)(smem + 24576);
  _Float16* ks = (_Float16*)(smem + 26624);
  _Float16* vs = (_Float16*)(smem + 28672);
  int head = blockIdx.x % NH_;
  int win  = blockIdx.x / NH_;
  int lane = threadIdx.x;
  const _Float16* qb = q  + (size_t)win * 64 * C_       + head * HD_;
  const _Float16* kb = kv + (size_t)win * 64 * (2 * C_) + head * HD_;
  const _Float16* vb = kb + C_;
#if HAVE_TDM
  tdm_load_2d((unsigned)(uintptr_t)qs, qb, 1, HD_, 64, C_,     HD_, 64);
  tdm_load_2d((unsigned)(uintptr_t)ks, kb, 1, HD_, 64, 2 * C_, HD_, 64);
  tdm_load_2d((unsigned)(uintptr_t)vs, vb, 1, HD_, 64, 2 * C_, HD_, 64);
  tdm_wait();
#else
  for (int i = lane; i < 64 * HD_; i += 32) {
    int r = i >> 4, c = i & 15;
    qs[i] = qb[r * C_ + c];
    ks[i] = kb[r * 2 * C_ + c];
    vs[i] = vb[r * 2 * C_ + c];
  }
  __syncthreads();
#endif

  // ---- S = q @ k^T  (hd=16 zero-padded to K=32) ----
  for (int ti = 0; ti < 4; ++ti) {
    v16h af;
    {
      int m = lane & 15;
      int kbA = (lane < 16) ? 0 : 16;
      const _Float16* ap = qs + (ti * 16 + m) * HD_;
#pragma unroll
      for (int v = 0; v < 8; ++v) {
        int k = kbA + 2 * v;
        af[2 * v]     = (k     < HD_) ? ap[k]     : (_Float16)0.f;
        af[2 * v + 1] = (k + 1 < HD_) ? ap[k + 1] : (_Float16)0.f;
      }
    }
    for (int tj = 0; tj < 4; ++tj) {
      v16h bfr;
      {
        int j = tj * 16 + (lane & 15);
        const _Float16* kp = ks + j * HD_;
        if (lane < 16) {
#pragma unroll
          for (int v = 0; v < 8; ++v) { bfr[2 * v] = kp[2 * v]; bfr[2 * v + 1] = kp[2 * v + 1]; }
        } else {
#pragma unroll
          for (int v = 0; v < 8; ++v) { bfr[2 * v] = (_Float16)0.f; bfr[2 * v + 1] = (_Float16)0.f; }
        }
      }
      v8f c = {};
      c = wmma32(af, bfr, c);
      int n = lane & 15, mb = (lane < 16) ? 0 : 8;
#pragma unroll
      for (int r = 0; r < 8; ++r)
        S[(ti * 16 + mb + r) * 64 + tj * 16 + n] = c[r];
    }
  }
  __syncthreads();

  // ---- bias + shift-mask + softmax (2 rows per lane) ----
  int wi = win & 255;
  int wh = wi >> 4, wc = wi & 15;
  for (int rr = 0; rr < 2; ++rr) {
    int i = lane * 2 + rr;
    int r1 = i >> 3, c1 = i & 7;
    int hh1 = wh * 8 + r1, ww1 = wc * 8 + c1;
    int reg1 = ((hh1 < 120) ? 0 : (hh1 < 124) ? 1 : 2) * 3 +
               ((ww1 < 120) ? 0 : (ww1 < 124) ? 1 : 2);
    float mx = -1e30f;
    for (int j = 0; j < 64; ++j) {
      int r2 = j >> 3, c2 = j & 7;
      int idx = (r1 - r2 + 7) * 15 + (c1 - c2 + 7);
      float v = S[i * 64 + j] + rpb[idx * NH_ + head];
      int hh2 = wh * 8 + r2, ww2 = wc * 8 + c2;
      int reg2 = ((hh2 < 120) ? 0 : (hh2 < 124) ? 1 : 2) * 3 +
                 ((ww2 < 120) ? 0 : (ww2 < 124) ? 1 : 2);
      if (reg1 != reg2) v -= 100.f;
      S[i * 64 + j] = v;
      mx = fmaxf(mx, v);
    }
    float sum = 0.f;
    for (int j = 0; j < 64; ++j) {
      float e = __expf(S[i * 64 + j] - mx);
      S[i * 64 + j] = e;
      sum += e;
    }
    float inv = 1.f / sum;
    for (int j = 0; j < 64; ++j)
      P[i * 64 + j] = (_Float16)(S[i * 64 + j] * inv);
  }
  __syncthreads();

  // ---- O = P @ V ----
  for (int ti = 0; ti < 4; ++ti) {
    v8f c = {};
    for (int k0 = 0; k0 < 64; k0 += 32) {
      v16h af, bfr;
      {
        int m = lane & 15;
        int kbA = (lane < 16) ? 0 : 16;
        const _Float16* pp = &P[(ti * 16 + m) * 64 + k0];
#pragma unroll
        for (int v = 0; v < 8; ++v) {
          int k = kbA + 2 * v;
          af[2 * v] = pp[k]; af[2 * v + 1] = pp[k + 1];
        }
      }
      {
        int n = lane & 15;
        int kbB = (lane < 16) ? 0 : 16;
#pragma unroll
        for (int v = 0; v < 8; ++v) {
          int kk = k0 + kbB + 2 * v;
          bfr[2 * v]     = vs[kk * HD_ + n];
          bfr[2 * v + 1] = vs[(kk + 1) * HD_ + n];
        }
      }
      c = wmma32(af, bfr, c);
    }
    int n = lane & 15, mb = (lane < 16) ? 0 : 8;
#pragma unroll
    for (int r = 0; r < 8; ++r)
      O[(size_t)(win * 64 + ti * 16 + mb + r) * C_ + head * HD_ + n] = (_Float16)c[r];
  }
}

// ---------------------------------------------------------------------------
// K4: proj GEMM strip + window-reverse + roll(+SS) + residual -> x1 (f32)
// ---------------------------------------------------------------------------
__global__ void k_proj_res(const _Float16* __restrict__ Aa,
                           const float* __restrict__ Wt,
                           const float* __restrict__ bias,
                           const float* __restrict__ xres,
                           float* __restrict__ x1) {
  constexpr int K = C_, N = C_;
  constexpr int WBYTES = K * 16 * 4;
  constexpr int ABYTES = 16 * K * 2;
  __shared__ __align__(16) char smem[WBYTES + 2 * ABYTES];
  float*    Ws = (float*)smem;
  _Float16* A0 = (_Float16*)(smem + WBYTES);
  _Float16* A1 = (_Float16*)(smem + WBYTES + ABYTES);
  int ncols = N >> 4;
  int tm0 = (blockIdx.x / ncols) * MSTRIP;
  int tn  = blockIdx.x % ncols;
  int lane = threadIdx.x;
  const int n  = (lane & 15) + tn * 16;
  const int mb = (lane < 16) ? 0 : 8;
  float bv = bias[n];
#if HAVE_TDM
  tdm_load_2d((unsigned)(uintptr_t)Ws, Wt + tn * 16, 2, N, K, N, 16, K);
  tdm_load_2d((unsigned)(uintptr_t)A0, Aa + (size_t)tm0 * 16 * K, 1, K, 16, K, K, 16);
  tdm_wait();
#else
  for (int i = lane; i < K * 16; i += 32)
    Ws[i] = Wt[(size_t)(i >> 4) * N + tn * 16 + (i & 15)];
  __syncthreads();
#endif
  v16h bf[K / 32];
  load_bfrag<K>(bf, Ws, lane);
  for (int i = 0; i < MSTRIP; ++i) {
#if HAVE_TDM
    const _Float16* Acur = (i & 1) ? A1 : A0;
    _Float16*       Anxt = (i & 1) ? A0 : A1;
    if (i + 1 < MSTRIP)
      tdm_load_2d((unsigned)(uintptr_t)Anxt,
                  Aa + (size_t)(tm0 + i + 1) * 16 * K, 1, K, 16, K, K, 16);
#else
    const _Float16* Acur = A0;
    for (int idx = lane; idx < 16 * K; idx += 32)
      A0[idx] = Aa[(size_t)(tm0 + i) * 16 * K + idx];
    __syncthreads();
#endif
    v8f acc = gemm_tile_reg<K>(Acur, bf, lane);
#pragma unroll
    for (int r = 0; r < 8; ++r) {
      int t = (tm0 + i) * 16 + mb + r;
      int win = t >> 6, tok = t & 63;
      int bt = win >> 8, wi = win & 255;
      int wh = wi >> 4, wc = wi & 15;
      int th = tok >> 3, tw = tok & 7;
      int h = (wh * 8 + th + SS_) & 127;
      int w = (wc * 8 + tw + SS_) & 127;
      size_t dst = ((size_t)bt * 16384 + h * 128 + w) * C_ + n;
      x1[dst] = xres[dst] + acc[r] + bv;
    }
#if HAVE_TDM
    tdm_wait();
#else
    __syncthreads();
#endif
  }
}

// ---------------------------------------------------------------------------
// K5: final MLP GEMM strip (f32 out) + residual (K compile-time = 384)
// ---------------------------------------------------------------------------
template <int K>
__global__ void k_gemm_out(const _Float16* __restrict__ Aa,
                           const float* __restrict__ Wt,
                           const float* __restrict__ bias,
                           const float* __restrict__ res,
                           float* __restrict__ O, int N) {
  constexpr int WBYTES = K * 16 * 4;
  constexpr int ABYTES = 16 * K * 2;
  __shared__ __align__(16) char smem[WBYTES + 2 * ABYTES];
  float*    Ws = (float*)smem;
  _Float16* A0 = (_Float16*)(smem + WBYTES);
  _Float16* A1 = (_Float16*)(smem + WBYTES + ABYTES);
  int ncols = N >> 4;
  int tm0 = (blockIdx.x / ncols) * MSTRIP;
  int tn  = blockIdx.x % ncols;
  int lane = threadIdx.x;
  const int n  = (lane & 15) + tn * 16;
  const int mb = (lane < 16) ? 0 : 8;
  float bv = bias[n];
#if HAVE_TDM
  tdm_load_2d((unsigned)(uintptr_t)Ws, Wt + tn * 16, 2, N, K, N, 16, K);
  tdm_load_2d((unsigned)(uintptr_t)A0, Aa + (size_t)tm0 * 16 * K, 1, K, 16, K, K, 16);
  tdm_wait();
#else
  for (int i = lane; i < K * 16; i += 32)
    Ws[i] = Wt[(size_t)(i >> 4) * N + tn * 16 + (i & 15)];
  __syncthreads();
#endif
  v16h bf[K / 32];
  load_bfrag<K>(bf, Ws, lane);
  for (int i = 0; i < MSTRIP; ++i) {
#if HAVE_TDM
    const _Float16* Acur = (i & 1) ? A1 : A0;
    _Float16*       Anxt = (i & 1) ? A0 : A1;
    if (i + 1 < MSTRIP)
      tdm_load_2d((unsigned)(uintptr_t)Anxt,
                  Aa + (size_t)(tm0 + i + 1) * 16 * K, 1, K, 16, K, K, 16);
#else
    const _Float16* Acur = A0;
    for (int idx = lane; idx < 16 * K; idx += 32)
      A0[idx] = Aa[(size_t)(tm0 + i) * 16 * K + idx];
    __syncthreads();
#endif
    v8f acc = gemm_tile_reg<K>(Acur, bf, lane);
#pragma unroll
    for (int r = 0; r < 8; ++r) {
      size_t idx = (size_t)((tm0 + i) * 16 + mb + r) * N + n;
      O[idx] = res[idx] + acc[r] + bv;
    }
#if HAVE_TDM
    tdm_wait();
#else
    __syncthreads();
#endif
  }
}

// ---------------------------------------------------------------------------
extern "C" void kernel_launch(void* const* d_in, const int* in_sizes, int n_in,
                              void* d_out, int out_size, void* d_ws, size_t ws_size,
                              hipStream_t stream) {
  const float* x    = (const float*)d_in[0];
  const float* y    = (const float*)d_in[1];
  const float* g1A  = (const float*)d_in[4];  const float* b1A = (const float*)d_in[5];
  const float* qAw  = (const float*)d_in[6];  const float* qAb = (const float*)d_in[7];
  const float* kvAw = (const float*)d_in[8];  const float* kvAb= (const float*)d_in[9];
  const float* pAw  = (const float*)d_in[10]; const float* pAb = (const float*)d_in[11];
  const float* rpbA = (const float*)d_in[12];
  const float* g2A  = (const float*)d_in[13]; const float* b2A = (const float*)d_in[14];
  const float* f1Aw = (const float*)d_in[15]; const float* f1Ab= (const float*)d_in[16];
  const float* f2Aw = (const float*)d_in[17]; const float* f2Ab= (const float*)d_in[18];
  const float* g1B  = (const float*)d_in[19]; const float* b1B = (const float*)d_in[20];
  const float* qBw  = (const float*)d_in[21]; const float* qBb = (const float*)d_in[22];
  const float* kvBw = (const float*)d_in[23]; const float* kvBb= (const float*)d_in[24];
  const float* pBw  = (const float*)d_in[25]; const float* pBb = (const float*)d_in[26];
  const float* rpbB = (const float*)d_in[27];
  const float* g2B  = (const float*)d_in[28]; const float* b2B = (const float*)d_in[29];
  const float* f1Bw = (const float*)d_in[30]; const float* f1Bb= (const float*)d_in[31];
  const float* f2Bw = (const float*)d_in[32]; const float* f2Bb= (const float*)d_in[33];

  const size_t S96h  = (size_t)MTOT * 96  * 2;
  const size_t S192h = (size_t)MTOT * 192 * 2;
  const size_t S96f  = (size_t)MTOT * 96  * 4;
  char* base = (char*)d_ws;
  _Float16* xw   = (_Float16*)(base);
  _Float16* yw   = (_Float16*)(base + 1 * S96h);
  _Float16* qA   = (_Float16*)(base + 2 * S96h);
  _Float16* qB   = (_Float16*)(base + 3 * S96h);
  _Float16* kvA  = (_Float16*)(base + 4 * S96h);
  _Float16* kvB  = (_Float16*)(base + 4 * S96h + S192h);
  _Float16* attn = (_Float16*)(base + 4 * S96h + 2 * S192h);
  float*    x1   = (float*)   (base + 5 * S96h + 2 * S192h);
  float*    y1   = (float*)   (base + 5 * S96h + 2 * S192h + S96f);
  _Float16* h16  = qA;    // reuse: q dead after attention
  _Float16* g16  = kvA;   // reuse: kvA+kvB span = MLP hidden, dead after proj
  float* outx = (float*)d_out;
  float* outy = outx + (size_t)MTOT * 96;

  const float qscale = 0.25f;  // hd^-0.5
  dim3 b256(256), b32(32);
  const int gLN   = MTOT / 256;
  const int NSTR  = MTOT / (16 * MSTRIP);        // 512 M-strips
  const int gS96  = NSTR * (96  / 16);           // 3072
  const int gS192 = NSTR * (192 / 16);           // 6144
  const int gS384 = NSTR * (384 / 16);           // 12288
  const int gAT   = NWINS * NH_;                 // 12288

  k_ln_window<<<gLN, b256, 0, stream>>>(x, g1A, b1A, xw);
  k_ln_window<<<gLN, b256, 0, stream>>>(y, g1B, b1B, yw);
  k_gemm_f16<96><<<gS96,  b32, 0, stream>>>(xw, qAw,  qAb,  qA,  96,  qscale, 0);
  k_gemm_f16<96><<<gS192, b32, 0, stream>>>(yw, kvAw, kvAb, kvA, 192, 1.f,    0);
  k_gemm_f16<96><<<gS96,  b32, 0, stream>>>(yw, qBw,  qBb,  qB,  96,  qscale, 0);
  k_gemm_f16<96><<<gS192, b32, 0, stream>>>(xw, kvBw, kvBb, kvB, 192, 1.f,    0);
  k_attn<<<gAT, b32, 0, stream>>>(qA, kvA, rpbA, attn);
  k_proj_res<<<gS96, b32, 0, stream>>>(attn, pAw, pAb, x, x1);
  k_attn<<<gAT, b32, 0, stream>>>(qB, kvB, rpbB, attn);
  k_proj_res<<<gS96, b32, 0, stream>>>(attn, pBw, pBb, y, y1);
  k_ln_plain<<<gLN, b256, 0, stream>>>(x1, g2A, b2A, h16);
  k_gemm_f16<96><<<gS384, b32, 0, stream>>>(h16, f1Aw, f1Ab, g16, 384, 1.f, 1);
  k_gemm_out<384><<<gS96, b32, 0, stream>>>(g16, f2Aw, f2Ab, x1, outx, 96);
  k_ln_plain<<<gLN, b256, 0, stream>>>(y1, g2B, b2B, h16);
  k_gemm_f16<96><<<gS384, b32, 0, stream>>>(h16, f1Bw, f1Bb, g16, 384, 1.f, 1);
  k_gemm_out<384><<<gS96, b32, 0, stream>>>(g16, f2Bw, f2Bb, y1, outy, 96);
}
